// TopK_6124623364323
// MI455X (gfx1250) — matrix operand order repaired and checked by reading, not compiled, not run on hardware
//
#include <hip/hip_runtime.h>

// Problem constants (from reference): N=8192, H=2048, L=16384, K=64
#define N_ROWS 8192
#define H_DIM  2048
#define L_DIM  16384
#define K_TOP  64

// GEMM tiling
#define BM 128
#define BN 128
#define BK 32
#define KSTEPS (H_DIM / BK)

typedef __attribute__((ext_vector_type(16))) __bf16 v16bf;
typedef __attribute__((ext_vector_type(8)))  __bf16 v8bf;
typedef __attribute__((ext_vector_type(8)))  float  v8f;

union FragU { v16bf v; v8bf h[2]; };

// ---- CDNA5 async global->LDS path (ASYNCcnt), guarded for toolchain support ----
#if __has_builtin(__builtin_amdgcn_global_load_async_to_lds_b128)
#define HAS_ASYNC_LDS 1
#else
#define HAS_ASYNC_LDS 0
#endif

typedef int vi4 __attribute__((vector_size(4 * sizeof(int))));
typedef __attribute__((address_space(1))) vi4* gbl_vi4_ptr;
typedef __attribute__((address_space(3))) vi4* lds_vi4_ptr;

__device__ __forceinline__ void wait_async_zero() {
#if HAS_ASYNC_LDS
#if __has_builtin(__builtin_amdgcn_s_wait_asynccnt)
  __builtin_amdgcn_s_wait_asynccnt(0);
#else
  asm volatile("s_wait_asynccnt 0x0" ::: "memory");
#endif
#endif
}

// Copy 16 bytes global -> LDS. Async variant writes straight to LDS (no VGPR
// round-trip) and is tracked by ASYNCcnt; fallback is load_b128 + ds_store_b128.
__device__ __forceinline__ void copy16_g2lds(unsigned short* dst_lds,
                                             const unsigned short* src_glb) {
#if HAS_ASYNC_LDS
  __builtin_amdgcn_global_load_async_to_lds_b128(
      (gbl_vi4_ptr)src_glb,
      (lds_vi4_ptr)dst_lds,
      /*offset=*/0, /*cpol=*/0);
#else
  *(uint4*)dst_lds = *(const uint4*)src_glb;
#endif
}

__device__ __forceinline__ unsigned short f2bf(float f) {
  unsigned u = __float_as_uint(f);
  u += 0x7FFFu + ((u >> 16) & 1u);       // round-to-nearest-even
  return (unsigned short)(u >> 16);
}

// Monotonic float -> uint key (larger float => larger key)
__device__ __forceinline__ unsigned keyOf(float f) {
  unsigned u = __float_as_uint(f);
  return (u & 0x80000000u) ? ~u : (u | 0x80000000u);
}

// ---------------- Kernel 1: xb = bf16(x - pre_bias) ----------------
__global__ __launch_bounds__(256) void prep_x(const float* __restrict__ x,
                                              const float* __restrict__ pre_bias,
                                              unsigned short* __restrict__ xb) {
  size_t stride = (size_t)gridDim.x * blockDim.x;
  size_t total  = (size_t)N_ROWS * H_DIM;
  for (size_t i = (size_t)blockIdx.x * blockDim.x + threadIdx.x; i < total; i += stride) {
    int h = (int)(i & (H_DIM - 1));
    xb[i] = f2bf(x[i] - pre_bias[h]);
  }
}

// ---------------- Kernel 2: wb = bf16(W_enc) ----------------
__global__ __launch_bounds__(256) void prep_w(const float* __restrict__ w,
                                              unsigned short* __restrict__ wb) {
  size_t stride = (size_t)gridDim.x * blockDim.x;
  size_t total  = (size_t)L_DIM * H_DIM;
  for (size_t i = (size_t)blockIdx.x * blockDim.x + threadIdx.x; i < total; i += stride) {
    wb[i] = f2bf(w[i]);
  }
}

// ---------------- Kernel 3: WdT[l][h] = W_dec[h][l] (tiled f32 transpose) ----------------
__global__ __launch_bounds__(256) void transpose_wdec(const float* __restrict__ Wd,
                                                      float* __restrict__ WdT) {
  __shared__ float tile[32][33];
  int l0 = blockIdx.x * 32;
  int h0 = blockIdx.y * 32;
  int tx = threadIdx.x & 31;
  int ty = threadIdx.x >> 5;   // 0..7
#pragma unroll
  for (int j = 0; j < 32; j += 8)
    tile[ty + j][tx] = Wd[(size_t)(h0 + ty + j) * L_DIM + (l0 + tx)];   // tile[h][l]
  __syncthreads();
#pragma unroll
  for (int j = 0; j < 32; j += 8)
    WdT[(size_t)(l0 + ty + j) * H_DIM + (h0 + tx)] = tile[tx][ty + j];
}

// ---------------- Kernel 4: encoder GEMM (bf16 WMMA, f32 acc) ----------------
// C[n,l] = sum_h xb[n,h] * wb[l,h] + latent_bias[l]  -> pre_acts (latents region)
// Double-buffered LDS pipeline, one barrier per K-step:
//   wait(my tile-t async loads) ; barrier ; issue tile t+1 ; compute tile t
__global__ __launch_bounds__(256) void encoder_gemm(const unsigned short* __restrict__ xb,
                                                    const unsigned short* __restrict__ wb,
                                                    const float* __restrict__ latent_bias,
                                                    float* __restrict__ pre_acts) {
  __shared__ __align__(16) unsigned short As[2][BM * BK];  // 2 x 8 KB
  __shared__ __align__(16) unsigned short Bs[2][BN * BK];  // 2 x 8 KB

  const int tid   = threadIdx.x;
  const int lane  = tid & 31;
  const int wid   = tid >> 5;       // 0..7
  const int waveM = wid >> 1;       // 0..3 : 32 rows of C each
  const int waveN = wid & 1;        // 0..1 : 64 cols of C each

  const int m_blk = blockIdx.y * BM;
  const int n_blk = blockIdx.x * BN;

  v8f acc[2][4];
#pragma unroll
  for (int im = 0; im < 2; ++im)
#pragma unroll
    for (int jn = 0; jn < 4; ++jn)
      acc[im][jn] = (v8f){0.f, 0.f, 0.f, 0.f, 0.f, 0.f, 0.f, 0.f};

  // staging: 256 threads x 2 chunks x 16B covers one 128x32 bf16 tile
  const int c1  = tid + 256;
  const int r0  = tid >> 2, k0c = (tid & 3) << 3;
  const int r1  = c1 >> 2,  k1c = (c1 & 3) << 3;

  const unsigned short* aRow0 = xb + (size_t)(m_blk + r0) * H_DIM + k0c;
  const unsigned short* aRow1 = xb + (size_t)(m_blk + r1) * H_DIM + k1c;
  const unsigned short* bRow0 = wb + (size_t)(n_blk + r0) * H_DIM + k0c;
  const unsigned short* bRow1 = wb + (size_t)(n_blk + r1) * H_DIM + k1c;

  auto stage = [&](int buf, int k0) {
    copy16_g2lds(&As[buf][r0 * BK + k0c], aRow0 + k0);
    copy16_g2lds(&As[buf][r1 * BK + k1c], aRow1 + k0);
    copy16_g2lds(&Bs[buf][r0 * BK + k0c], bRow0 + k0);
    copy16_g2lds(&Bs[buf][r1 * BK + k1c], bRow1 + k0);
  };

  stage(0, 0);   // prologue: tile 0 in flight

  for (int t = 0; t < KSTEPS; ++t) {
    const int cur = t & 1;
    wait_async_zero();          // my tile-t writes have landed in LDS
    __syncthreads();            // everyone's tile-t data visible; prev reads done

    if (t + 1 < KSTEPS) stage(cur ^ 1, (t + 1) * BK);

    // A fragment layout (ISA 7.12.2, 16-bit A 16x32): lanes 0-15 hold M=lane,
    // K{0..7} in v0-3 and K{16..23} in v4-7; lanes 16-31 hold K{8..15},{24..31}.
    FragU a[2], b[4];
    {
      const int m  = lane & 15;
      const int ka = (lane >> 4) * 8;
#pragma unroll
      for (int im = 0; im < 2; ++im) {
        const int row = waveM * 32 + im * 16 + m;
        a[im].h[0] = *(const v8bf*)&As[cur][row * BK + ka];
        a[im].h[1] = *(const v8bf*)&As[cur][row * BK + ka + 16];
      }
      // B fragment (32x16, 16-bit): lane n holds column N=lane&15;
      // lanes 0-15: K{0..7},{16..23}, lanes 16-31: K{8..15},{24..31}, 2 packed per VGPR.
      const int nn = lane & 15;
      const int kb = (lane >> 4) * 8;
#pragma unroll
      for (int jn = 0; jn < 4; ++jn) {
        const int row = waveN * 64 + jn * 16 + nn;
        b[jn].h[0] = *(const v8bf*)&Bs[cur][row * BK + kb];
        b[jn].h[1] = *(const v8bf*)&Bs[cur][row * BK + kb + 16];
      }
    }
#pragma unroll
    for (int im = 0; im < 2; ++im)
#pragma unroll
      for (int jn = 0; jn < 4; ++jn)
        acc[im][jn] = __builtin_amdgcn_wmma_f32_16x16x32_bf16(
            false, a[im].v, false, b[jn].v, (short)0, acc[im][jn], false, false);
  }

  // Epilogue: C 16x16 f32 layout — lane n = lane&15, VGPR r -> M = r + 8*(lane>>4)
  const int n_lane = lane & 15;
  const int m_half = (lane >> 4) * 8;
#pragma unroll
  for (int jn = 0; jn < 4; ++jn) {
    const int ng = n_blk + waveN * 64 + jn * 16 + n_lane;
    const float lb = latent_bias[ng];
#pragma unroll
    for (int im = 0; im < 2; ++im) {
      const int mg0 = m_blk + waveM * 32 + im * 16 + m_half;
#pragma unroll
      for (int r = 0; r < 8; ++r)
        pre_acts[(size_t)(mg0 + r) * L_DIM + ng] = acc[im][jn][r] + lb;
    }
  }
}

// ---------------- Kernel 5: per-row top-k=64 radix select (in place) ----------------
__global__ __launch_bounds__(256) void topk_row(float* __restrict__ latents,
                                                float* __restrict__ vals,
                                                int* __restrict__ idxs) {
  const int row = blockIdx.x;
  const int tid = threadIdx.x;
  float* rp = latents + (size_t)row * L_DIM;

  __shared__ unsigned hist[256];
  __shared__ unsigned s_prefix;
  __shared__ int s_need;
  __shared__ unsigned s_sel, s_eq;

  if (tid == 0) { s_prefix = 0u; s_need = K_TOP; }
  __syncthreads();

  for (int shift = 24; shift >= 0; shift -= 8) {
    hist[tid] = 0u;
    __syncthreads();
    const unsigned prefix = s_prefix;
    const unsigned pmask  = (shift == 24) ? 0u : (0xFFFFFFFFu << (shift + 8));
    for (int i = tid; i < L_DIM; i += 256) {
      unsigned key = keyOf(rp[i]);
      if ((key & pmask) == prefix)
        atomicAdd(&hist[(key >> shift) & 0xFF], 1u);
    }
    __syncthreads();
    if (tid == 0) {
      int need = s_need;
      unsigned cum = 0;
      int b = 255;
      for (; b > 0; --b) {
        unsigned c = hist[b];
        if ((int)(cum + c) >= need) break;
        cum += c;
      }
      s_prefix = prefix | ((unsigned)b << shift);
      s_need   = need - (int)cum;
    }
    __syncthreads();
  }

  const unsigned T = s_prefix;     // exact key of the k-th largest element
  const int need_eq = s_need;      // how many ==T to keep
  if (tid == 0) { s_sel = 0u; s_eq = 0u; }
  __syncthreads();

  float* vrow = vals + (size_t)row * K_TOP;
  int*   irow = idxs + (size_t)row * K_TOP;

  for (int i = tid; i < L_DIM; i += 256) {
    const float v = rp[i];
    const unsigned key = keyOf(v);
    float outv = 0.0f;
    if (key > T) {
      unsigned slot = atomicAdd(&s_sel, 1u);
      vrow[slot] = v; irow[slot] = i;
      outv = v;
    } else if (key == T) {
      unsigned e = atomicAdd(&s_eq, 1u);
      if ((int)e < need_eq) {
        unsigned slot = atomicAdd(&s_sel, 1u);
        vrow[slot] = v; irow[slot] = i;
        outv = v;
      }
    }
    rp[i] = outv;   // zero everything not selected; keep top-k values
  }
}

// ---------------- Kernel 6: sparse decode x_hat[n] = pre_bias + sum_j v_j * WdT[idx_j,:] ----
__global__ __launch_bounds__(256) void decoder(const float* __restrict__ vals,
                                               const int* __restrict__ idxs,
                                               const float* __restrict__ WdT,
                                               const float* __restrict__ pre_bias,
                                               float* __restrict__ xhat) {
  const int row = blockIdx.x;
  const int tid = threadIdx.x;
  __shared__ float sv[K_TOP];
  __shared__ int   si[K_TOP];
  if (tid < K_TOP) {
    sv[tid] = vals[(size_t)row * K_TOP + tid];
    si[tid] = idxs[(size_t)row * K_TOP + tid];
  }
  __syncthreads();

  float acc[H_DIM / 256];
#pragma unroll
  for (int c = 0; c < H_DIM / 256; ++c) acc[c] = pre_bias[tid + 256 * c];

  for (int j = 0; j < K_TOP; ++j) {
    const float* wr = WdT + (size_t)si[j] * H_DIM;
    const float s = sv[j];
#pragma unroll
    for (int c = 0; c < H_DIM / 256; ++c)
      acc[c] = fmaf(s, wr[tid + 256 * c], acc[c]);
  }
  float* orow = xhat + (size_t)row * H_DIM;
#pragma unroll
  for (int c = 0; c < H_DIM / 256; ++c) orow[tid + 256 * c] = acc[c];
}

// ---------------- Launcher ----------------
extern "C" void kernel_launch(void* const* d_in, const int* in_sizes, int n_in,
                              void* d_out, int out_size, void* d_ws, size_t ws_size,
                              hipStream_t stream) {
  const float* x           = (const float*)d_in[0];   // [N,H]
  const float* W_enc       = (const float*)d_in[1];   // [L,H]
  const float* W_dec       = (const float*)d_in[2];   // [H,L]
  const float* pre_bias    = (const float*)d_in[3];   // [H]
  const float* latent_bias = (const float*)d_in[4];   // [L]
  (void)in_sizes; (void)n_in; (void)out_size; (void)ws_size;

  float* latents = (float*)d_out;                     // [N,L]
  float* xhat    = latents + (size_t)N_ROWS * L_DIM;  // [N,H]

  // Workspace layout (~228 MB): xb | wb | WdT | vals | idxs
  char* ws = (char*)d_ws;
  unsigned short* xb  = (unsigned short*)ws;                       // N*H bf16   (32 MB)
  unsigned short* wbf = xb + (size_t)N_ROWS * H_DIM;               // L*H bf16   (64 MB)
  float* WdT  = (float*)(wbf + (size_t)L_DIM * H_DIM);             // L*H f32    (128 MB)
  float* vals = WdT + (size_t)L_DIM * H_DIM;                       // N*64 f32   (2 MB)
  int*   idxs = (int*)(vals + (size_t)N_ROWS * K_TOP);             // N*64 i32   (2 MB)

  prep_x<<<2048, 256, 0, stream>>>(x, pre_bias, xb);
  prep_w<<<4096, 256, 0, stream>>>(W_enc, wbf);

  dim3 tgrid(L_DIM / 32, H_DIM / 32);
  transpose_wdec<<<tgrid, 256, 0, stream>>>(W_dec, WdT);

  dim3 ggrid(L_DIM / BN, N_ROWS / BM);   // (128, 64) blocks, 256 threads (8 waves) each
  encoder_gemm<<<ggrid, 256, 0, stream>>>(xb, wbf, latent_bias, latents);

  topk_row<<<N_ROWS, 256, 0, stream>>>(latents, vals, idxs);
  decoder<<<N_ROWS, 256, 0, stream>>>(vals, idxs, WdT, pre_bias, xhat);
}
